// CIF_55284819034390
// MI455X (gfx1250) — matrix operand order repaired and checked by reading, not compiled, not run on hardware
//
#include <hip/hip_runtime.h>
#include <hip/hip_bf16.h>

#define BATCH 16
#define S_LEN 1500
#define CCH   768
#define XP    1506      // padded rows per batch: 1 zero front, 5 zero back
#define T_MAX 75
#define KTOT  2304      // 3*768 contraction length
#define KTILES 72       // KTOT/32
#define CTILES 48       // 768/16
#define STILES32 47     // ceil(1500/32) row-tiles of 32
#define GWAVES 12       // waves per gemm block
#define GTPT 4          // col-tiles per wave (12*4 = 48)

typedef __attribute__((ext_vector_type(16))) __bf16 bf16x16;
typedef __attribute__((ext_vector_type(8)))  float  v8f;

__device__ __forceinline__ unsigned short f2bf(float f) {
    unsigned int u = __float_as_uint(f);
    unsigned int r = u + 0x7fffu + ((u >> 16) & 1u);   // round to nearest even
    return (unsigned short)(r >> 16);
}

// ---------------------------------------------------------------------------
// Pack conv_w (O=768, I=768, K=3) f32 into WMMA B fragments (bf16).
// Fragment (ct, kc): lane L holds column n = ct*16 + (L&15),
// K-values kbase..kbase+7, kbase+16..kbase+23 with kbase = kc*32 + (L>>4)*8, in 32B.
// Contraction index kg = k3*768 + i  => value = w[col, i, k3].
// ---------------------------------------------------------------------------
__global__ void pack_w_kernel(const float* __restrict__ conv_w,
                              unsigned short* __restrict__ bpack) {
    int tid = blockIdx.x * 256 + threadIdx.x;
    if (tid >= CTILES * KTILES * 32) return;
    int lane = tid & 31;
    int frag = tid >> 5;
    int kc = frag % KTILES;
    int ct = frag / KTILES;
    int col = ct * 16 + (lane & 15);
    int kbase = kc * 32 + (lane >> 4) * 8;
    unsigned short* dst = bpack + (size_t)frag * 512 + lane * 16;
    unsigned short tmp[16];
#pragma unroll
    for (int j = 0; j < 16; ++j) {
        int kg = kbase + j + (j >= 8 ? 8 : 0);
        int k3 = kg / 768;
        int i  = kg - k3 * 768;
        tmp[j] = f2bf(conv_w[(size_t)col * 2304 + i * 3 + k3]);
    }
    uint4* d4 = (uint4*)dst;
    d4[0] = ((uint4*)tmp)[0];
    d4[1] = ((uint4*)tmp)[1];
}

// ---------------------------------------------------------------------------
// Convert audio_feat f32 -> bf16 with zero halo rows (p=0 and p>=1501).
// xpad[b][p][i] = x[b][p-1][i] for p in 1..1500.
// ---------------------------------------------------------------------------
__global__ void pad_x_kernel(const float* __restrict__ x,
                             unsigned short* __restrict__ xpad) {
    long idx = (long)blockIdx.x * 256 + threadIdx.x;          // uint4 chunk id
    const long total = (long)BATCH * XP * (CCH / 8);
    if (idx >= total) return;
    int  chunk = (int)(idx % (CCH / 8));
    long row   = idx / (CCH / 8);
    int  p = (int)(row % XP);
    int  b = (int)(row / XP);
    uint4* dst = (uint4*)(xpad + idx * 8);
    if (p == 0 || p > 1500) {
        *dst = make_uint4(0u, 0u, 0u, 0u);
    } else {
        const float* src = x + ((size_t)b * S_LEN + (p - 1)) * CCH + chunk * 8;
        unsigned short tmp[8];
#pragma unroll
        for (int j = 0; j < 8; ++j) tmp[j] = f2bf(src[j]);
        *dst = *(uint4*)tmp;
    }
}

// ---------------------------------------------------------------------------
// Fused conv-GEMM (bf16 WMMA) + bias + ReLU + projection + sigmoid -> alpha.
// One block per 32-row s-tile: 12 waves x 4 col-tiles x 2 row-tiles.
// Each B fragment is shared by the top and bottom A fragments -> halves
// B traffic (2.65 GB L2/call) and gives 16 wmma : 16 global_load per 2 kc.
// Software-pipelined 2 deep; A via one LDS base (+24576 imm for bottom tile),
// B via one global base with t*73728 immediates.
// ---------------------------------------------------------------------------
union AF  { uint4 q[2]; bf16x16 v; };

__global__ void __launch_bounds__(384) gemm_alpha_kernel(
        const unsigned short* __restrict__ xpad,
        const unsigned short* __restrict__ bpack,
        const float* __restrict__ conv_b,
        const float* __restrict__ proj_w,
        const float* __restrict__ proj_b,
        const unsigned char* __restrict__ padmask,
        float* __restrict__ alpha_raw) {
    __shared__ unsigned short ldsX[34 * CCH];       // 52224 B (rows s0..s0+33)
    __shared__ float ldsP[384 * 16];                // 24576 B

    int b    = blockIdx.x / STILES32;
    int tile = blockIdx.x % STILES32;
    int s0   = tile * 32;

    // cooperative load of 34 padded rows (s0 .. s0+33) into LDS
    const uint4* src = (const uint4*)(xpad + ((size_t)b * XP + s0) * CCH);
    uint4* dst = (uint4*)ldsX;
    for (int i = threadIdx.x; i < 34 * CCH / 8; i += 384) dst[i] = src[i];
    __syncthreads();

    int lane = threadIdx.x & 31;
    int wave = threadIdx.x >> 5;
    int h    = lane >> 4;       // K-half
    int mrow = lane & 15;       // A row (within 16) / C col

    // A: byte addr = mrow*1536 + h*16 + kc*64 (+32 2nd K-run; +24576 bottom tile)
    const char* abase = (const char*)ldsX + mrow * 1536 + h * 16;
    // B: byte addr = wave*4*73728 + lane*32 + t*73728 + kc*1024 (+16)
    const char* bp = (const char*)bpack +
                     (size_t)wave * GTPT * (KTILES * 1024) + lane * 32;

    v8f accT[GTPT] = {};
    v8f accB[GTPT] = {};
    AF afT0, afB0, afT1, afB1;
    AF bv0[GTPT], bv1[GTPT];

#define LOADAB(AT_, AB_, BV_, KC_) do {                                     \
        const char* ap_ = abase + (KC_) * 64;                               \
        (AT_).q[0] = *(const uint4*)(ap_);                                  \
        (AT_).q[1] = *(const uint4*)(ap_ + 32);                             \
        (AB_).q[0] = *(const uint4*)(ap_ + 24576);                          \
        (AB_).q[1] = *(const uint4*)(ap_ + 24576 + 32);                     \
        const char* bq_ = bp + (size_t)(KC_) * 1024;                        \
        _Pragma("unroll")                                                   \
        for (int t_ = 0; t_ < GTPT; ++t_) {                                 \
            (BV_)[t_].q[0] = *(const uint4*)(bq_ + t_ * (KTILES * 1024));   \
            (BV_)[t_].q[1] = *(const uint4*)(bq_ + t_ * (KTILES * 1024) + 16); \
        }                                                                   \
    } while (0)

#define WMMA8(AT_, AB_, BV_) do {                                           \
        _Pragma("unroll")                                                   \
        for (int t_ = 0; t_ < GTPT; ++t_) {                                 \
            accT[t_] = __builtin_amdgcn_wmma_f32_16x16x32_bf16(             \
                false, (AT_).v, false, (BV_)[t_].v, (short)0,               \
                accT[t_], false, false);                                    \
            accB[t_] = __builtin_amdgcn_wmma_f32_16x16x32_bf16(             \
                false, (AB_).v, false, (BV_)[t_].v, (short)0,               \
                accB[t_], false, false);                                    \
        }                                                                   \
    } while (0)

    LOADAB(afT0, afB0, bv0, 0);
    for (int kc = 0; kc < KTILES; kc += 2) {
        LOADAB(afT1, afB1, bv1, kc + 1);    // prefetch odd step
        WMMA8(afT0, afB0, bv0);
        LOADAB(afT0, afB0, bv0, kc + 2);    // prefetch next even (kc=70 -> junk)
        WMMA8(afT1, afB1, bv1);
    }
#undef LOADAB
#undef WMMA8

    // epilogue: bias + relu + dot with proj_w, fold to per-row partials
    float p[16];
#pragma unroll
    for (int r = 0; r < 16; ++r) p[r] = 0.0f;
#pragma unroll
    for (int t = 0; t < GTPT; ++t) {
        int col = (wave * GTPT + t) * 16 + mrow;
        float bias = conv_b[col];
        float pw   = proj_w[col];
#pragma unroll
        for (int r = 0; r < 8; ++r) {                // row m = r + h*8 (+16 bottom)
            float hvT = accT[t][r] + bias;
            hvT = hvT > 0.0f ? hvT : 0.0f;
            p[r] += hvT * pw;
            float hvB = accB[t][r] + bias;
            hvB = hvB > 0.0f ? hvB : 0.0f;
            p[8 + r] += hvB * pw;
        }
    }
#pragma unroll
    for (int r = 0; r < 16; ++r) ldsP[threadIdx.x * 16 + r] = p[r];
    __syncthreads();

    if (threadIdx.x < 32) {
        int m  = threadIdx.x;        // row within 32-row tile
        int g  = m >> 4;             // 0 = top tile, 1 = bottom tile
        int mm = m & 15;
        int hh = mm >> 3, rr = mm & 7;
        float tot = 0.0f;
        for (int w = 0; w < GWAVES; ++w)
            for (int l = 0; l < 16; ++l)
                tot += ldsP[(w * 32 + hh * 16 + l) * 16 + g * 8 + rr];
        int s = s0 + m;
        if (s < S_LEN) {
            float z = tot + proj_b[0];
            float sig = 1.0f / (1.0f + __expf(-z));
            sig = sig < 0.0f ? 0.0f : (sig > 1.0f ? 1.0f : sig);
            if (padmask[(size_t)b * S_LEN + s]) sig = 0.0f;
            alpha_raw[b * S_LEN + s] = sig;
        }
    }
}

// ---------------------------------------------------------------------------
// Per-batch: sum, scale, cumsum, fire indices/weights, feat_lengths, masks.
// One block per batch; fixed-order reductions (deterministic).
// meta[s] = {left_idx, right_idx, right_weight(bits), left_weight(bits)}.
// ---------------------------------------------------------------------------
__global__ void cif_scan_kernel(const float* __restrict__ alpha_raw,
                                const int* __restrict__ target_lengths,
                                float* __restrict__ alpha_out,
                                float* __restrict__ fire_out,
                                float* __restrict__ feat_out,
                                float* __restrict__ pad_out,
                                int4* __restrict__ meta) {
    int b = blockIdx.x;
    int t = threadIdx.x;
    __shared__ float tsum[256];
    __shared__ float tpre[256];
    __shared__ float s_scale;
    __shared__ int   s_fl;

    const float* ar = alpha_raw + (size_t)b * S_LEN;
    float v[6];
    float loc = 0.0f;
#pragma unroll
    for (int j = 0; j < 6; ++j) {
        int s = t * 6 + j;
        float a = (s < S_LEN) ? ar[s] : 0.0f;
        v[j] = a; loc += a;
    }
    tsum[t] = loc;
    __syncthreads();
    if (t == 0) {
        float run = 0.0f;
        for (int i = 0; i < 256; ++i) { tpre[i] = run; run += tsum[i]; }
        float desired = (float)target_lengths[b] + 1e-5f;
        float scale   = desired / run;
        s_scale = scale;
        int fl = (int)floorf(run * scale);
        fl = fl < 1 ? 1 : (fl > T_MAX ? T_MAX : fl);
        s_fl = fl;
    }
    __syncthreads();
    float scale = s_scale;
    float pre = tpre[t];
    float cprev = pre * scale;
    int left = (int)floorf(cprev);
    left = left < 0 ? 0 : (left > T_MAX ? T_MAX : left);
    float run = pre;
#pragma unroll
    for (int j = 0; j < 6; ++j) {
        int s = t * 6 + j;
        if (s >= S_LEN) break;
        run += v[j];
        float csum = run * scale;
        float a = v[j] * scale;
        int right = (int)floorf(csum);
        right = right < 0 ? 0 : (right > T_MAX ? T_MAX : right);
        int fn = right - left;
        float rw = (fn > 0) ? (csum - (float)right) : 0.0f;
        int extra = fn > 1 ? fn - 1 : 0;
        float lw = a - rw - (float)extra;
        alpha_out[(size_t)b * S_LEN + s] = a;
        fire_out [(size_t)b * S_LEN + s] = (fn > 0) ? 1.0f : 0.0f;
        int4 m;
        m.x = left; m.y = right;
        m.z = __float_as_int(rw); m.w = __float_as_int(lw);
        meta[(size_t)b * S_LEN + s] = m;
        left = right;
    }
    if (t == 0) feat_out[b] = (float)s_fl;
    if (t < T_MAX) pad_out[b * T_MAX + t] = (t >= s_fl) ? 1.0f : 0.0f;
}

// ---------------------------------------------------------------------------
// Deterministic scatter as a gather: block = (b, t<75), threads = channels.
// Weight w(s,t) is wave-uniform -> scalar branch skips x loads when zero.
// ---------------------------------------------------------------------------
__global__ void __launch_bounds__(256) cif_scatter_kernel(
        const float* __restrict__ x,
        const int4* __restrict__ meta,
        float* __restrict__ out) {
    int b = blockIdx.x / T_MAX;
    int t = blockIdx.x % T_MAX;
    int c = threadIdx.x;
    const float* xb = x + (size_t)b * S_LEN * CCH;
    const int4* mb  = meta + (size_t)b * S_LEN;
    float a0 = 0.0f, a1 = 0.0f, a2 = 0.0f;
    for (int s = 0; s < S_LEN; ++s) {
        int4 m = mb[s];
        float w = 0.0f;
        if (m.y == t) w += __int_as_float(m.z);      // right_weight
        if (m.x == t) w += __int_as_float(m.w);      // left_weight
        int d = t - m.x;
        int extra = m.y - m.x - 1;
        if (d >= 1 && d <= 4 && extra >= d) w += 1.0f;   // THRESH * x
        w = __uint_as_float(__builtin_amdgcn_readfirstlane(__float_as_uint(w)));
        if (w != 0.0f) {
            const float* xs = xb + (size_t)s * CCH;
            a0 += w * xs[c];
            a1 += w * xs[c + 256];
            a2 += w * xs[c + 512];
        }
    }
    float* ob = out + ((size_t)b * T_MAX + t) * CCH;
    ob[c] = a0; ob[c + 256] = a1; ob[c + 512] = a2;
}

// ---------------------------------------------------------------------------
extern "C" void kernel_launch(void* const* d_in, const int* in_sizes, int n_in,
                              void* d_out, int out_size, void* d_ws, size_t ws_size,
                              hipStream_t stream) {
    const float*         audio  = (const float*)d_in[0];
    const unsigned char* mask   = (const unsigned char*)d_in[1];
    const int*           tlen   = (const int*)d_in[2];
    const float*         conv_w = (const float*)d_in[3];
    const float*         conv_b = (const float*)d_in[4];
    const float*         proj_w = (const float*)d_in[5];
    const float*         proj_b = (const float*)d_in[6];

    char* ws = (char*)d_ws;
    const size_t BPACK_BYTES = (size_t)CTILES * KTILES * 512 * 2;       // 3,538,944
    const size_t XPAD_BYTES  = (size_t)BATCH * XP * CCH * 2;            // 37,011,456
    const size_t ARAW_BYTES  = (size_t)BATCH * S_LEN * 4;               //     96,000
    unsigned short* bpack     = (unsigned short*)ws;
    unsigned short* xpad      = (unsigned short*)(ws + BPACK_BYTES);
    float*          alpha_raw = (float*)(ws + BPACK_BYTES + XPAD_BYTES);
    int4*           meta      = (int4*)(ws + BPACK_BYTES + XPAD_BYTES + ARAW_BYTES);

    float* out       = (float*)d_out;                   // (16,75,768)
    float* feat_out  = out + (size_t)BATCH * T_MAX * CCH;
    float* alpha_out = feat_out + BATCH;
    float* fire_out  = alpha_out + (size_t)BATCH * S_LEN;
    float* pad_out   = fire_out  + (size_t)BATCH * S_LEN;

    int packThreads = CTILES * KTILES * 32;                       // 110592
    pack_w_kernel<<<(packThreads + 255) / 256, 256, 0, stream>>>(conv_w, bpack);

    long chunks = (long)BATCH * XP * (CCH / 8);
    pad_x_kernel<<<(int)((chunks + 255) / 256), 256, 0, stream>>>(audio, xpad);

    gemm_alpha_kernel<<<BATCH * STILES32, 384, 0, stream>>>(
        xpad, bpack, conv_b, proj_w, proj_b, mask, alpha_raw);

    cif_scan_kernel<<<BATCH, 256, 0, stream>>>(
        alpha_raw, tlen, alpha_out, fire_out, feat_out, pad_out, meta);

    cif_scatter_kernel<<<BATCH * T_MAX, 256, 0, stream>>>(audio, meta, out);
}
